// DiMap_1_34832184770679
// MI455X (gfx1250) — compile-verified, matrix operand order checked by reading, not run on hardware
//
#include <hip/hip_runtime.h>
#include <math.h>

// ---------------- configuration ----------------
#define TS 66                 // LDS row stride (floats) for f32 matrices
#define MSZ (64 * TS)         // padded f32 matrix footprint in floats
#define NTHREADS 512          // 16 waves (wave32) -> 4x4 grid of 16x16 WMMA tiles
#define REDOFF (8 * MSZ)      // reduction scratch offset (512 floats)
#define HS 72                 // f16 staging row stride in halves (36 dwords: conflict-free rows)
#define A16_OFF ((8 * MSZ + NTHREADS) * 2)   // half-offsets into the LDS pool
#define B16_OFF (A16_OFF + 64 * HS)
#define POOL_FLOATS (8 * MSZ + NTHREADS + 64 * HS)  // f32 pool + red + 2 f16 buffers (floats)
#define EPS_REEIG 1e-4f
#define NS_ITERS 14           // Newton-Schulz iterations for sqrt/invsqrt
#define LOG_SQRTS 4           // inverse scaling-and-squaring depth for logm
#define LOG_TERMS 8           // Gregory/Taylor terms for log(I+M)
#define EXP_TERMS 8           // Taylor terms for expm
#define EXP_SQUARES 6         // squaring steps for expm (scale = 1/64)

// workspace layout (float offsets)
#define POOLED_OFF ((size_t)0)
#define LOGS_OFF   ((size_t)8388608)
#define G_OFF      ((size_t)16777216)
#define GS_OFF     (G_OFF + 4096)
#define GIS_OFF    (G_OFF + 8192)
#define GT_OFF     (G_OFF + 12288)
#define MIS_OFF    (G_OFF + 16384)
#define WSQ_OFF    (G_OFF + 20480)

typedef __attribute__((ext_vector_type(16))) _Float16 v16h;
typedef __attribute__((ext_vector_type(8)))  float    v8f;

// Synchronization contract: every block-level helper ENDS with __syncthreads().
// Helpers therefore do not need an entry barrier as long as they are only
// called after another helper (or an explicit __syncthreads()).

// ---------------- 64x64 GEMM via v_wmma_f32_16x16x32_f16 ----------------
// smem[dO..] = smem[aO..] * smem[bO..]  (f32 TS-layout 64x64).
// Phase 1: stage A -> packed f16 row-major (A16), B -> packed f16 TRANSPOSED
//          row-major (B16T); one conversion per element per GEMM.
// Phase 2: WMMA with two contiguous 8-half runs per fragment (b64-merged ds loads),
//          zero conversions on the read side. D may alias A/B.
__device__ __forceinline__ void mm64(int dO, int aO, int bO) {
  extern __shared__ float smem[];
  _Float16* hm = (_Float16*)smem;
  const int tid = threadIdx.x;
  // ---- stage operands (each thread: 8 consecutive elements of one row) ----
  {
    const int e0 = tid * 8;          // 512 threads x 8 = 4096 elements
    const int r  = e0 >> 6, c0 = e0 & 63;
#pragma unroll
    for (int j = 0; j < 8; ++j) {
      const int c = c0 + j;
      hm[A16_OFF + r * HS + c] = (_Float16)smem[aO + r * TS + c];  // A row-major
      hm[B16_OFF + c * HS + r] = (_Float16)smem[bO + r * TS + c];  // B transposed
    }
  }
  __syncthreads();
  // ---- WMMA phase ----
  const int wave = tid >> 5;
  const int lane = tid & 31;
  const int tm = (wave >> 2) << 4;
  const int tn = (wave &  3) << 4;
  const int lr = lane & 15;
  const int hi = lane >> 4;
  const int ar = A16_OFF + (tm + lr) * HS;   // A: one row per lane
  const int br = B16_OFF + (tn + lr) * HS;   // B^T: one (original) column per lane
  v8f acc = {0.f, 0.f, 0.f, 0.f, 0.f, 0.f, 0.f, 0.f};
#pragma unroll
  for (int kk = 0; kk < 64; kk += 32) {
    const int kb = kk + hi * 8;              // lane-group K split per ISA layout
    v16h a, b;
#pragma unroll
    for (int i = 0; i < 8; ++i) {            // two contiguous 16-byte runs each
      a[i]     = hm[ar + kb + i];
      a[8 + i] = hm[ar + kb + 16 + i];
      b[i]     = hm[br + kb + i];
      b[8 + i] = hm[br + kb + 16 + i];
    }
    acc = __builtin_amdgcn_wmma_f32_16x16x32_f16(false, a, false, b,
                                                 (short)0, acc, false, false);
  }
  const int dbase = dO + (tm + (hi << 3)) * TS + tn + lr;
#pragma unroll
  for (int r = 0; r < 8; ++r) smem[dbase + r * TS] = acc[r];
  __syncthreads();
}

// ---------------- elementwise helpers ----------------
// Each thread owns 8 consecutive elements of one row: fully unrolled,
// no exec-mask loop, b64-mergeable LDS and b128-mergeable global access.
#define EW_HEAD \
  extern __shared__ float smem[]; \
  const int e0 = threadIdx.x * 8;  \
  const int r = e0 >> 6, c0 = e0 & 63;

__device__ __forceinline__ void ew_scale_add_I(int dO, int sO, float a, float b) {
  EW_HEAD
#pragma unroll
  for (int j = 0; j < 8; ++j) {
    const int c = c0 + j;
    smem[dO + r * TS + c] = a * smem[sO + r * TS + c] + ((r == c) ? b : 0.f);
  }
  __syncthreads();
}
__device__ __forceinline__ void ew_setI(int dO) {
  EW_HEAD
#pragma unroll
  for (int j = 0; j < 8; ++j) {
    const int c = c0 + j;
    smem[dO + r * TS + c] = (r == c) ? 1.f : 0.f;
  }
  __syncthreads();
}
__device__ __forceinline__ void ew_zero(int dO) {
  EW_HEAD
#pragma unroll
  for (int j = 0; j < 8; ++j) smem[dO + r * TS + c0 + j] = 0.f;
  __syncthreads();
}
__device__ __forceinline__ void ew_copy(int dO, int sO) {
  EW_HEAD
#pragma unroll
  for (int j = 0; j < 8; ++j) smem[dO + r * TS + c0 + j] = smem[sO + r * TS + c0 + j];
  __syncthreads();
}
__device__ __forceinline__ void ew_axpy(int dO, int sO, float a) {
  EW_HEAD
#pragma unroll
  for (int j = 0; j < 8; ++j) smem[dO + r * TS + c0 + j] += a * smem[sO + r * TS + c0 + j];
  __syncthreads();
}
__device__ __forceinline__ void ew_load(int dO, const float* __restrict__ g) {
  EW_HEAD
#pragma unroll
  for (int j = 0; j < 8; ++j) smem[dO + r * TS + c0 + j] = g[e0 + j];
  __syncthreads();
}
__device__ __forceinline__ void ew_store(float* __restrict__ g, int sO) {
  EW_HEAD
#pragma unroll
  for (int j = 0; j < 8; ++j) g[e0 + j] = smem[sO + r * TS + c0 + j];
  __syncthreads();
}

// Frobenius norm of a TS-layout matrix (deterministic tree reduction).
__device__ __forceinline__ float frob(int aO) {
  EW_HEAD
  float p = 0.f;
#pragma unroll
  for (int j = 0; j < 8; ++j) {
    const float v = smem[aO + r * TS + c0 + j];
    p += v * v;
  }
  smem[REDOFF + threadIdx.x] = p;
  __syncthreads();
  for (int s = NTHREADS / 2; s > 0; s >>= 1) {
    if (threadIdx.x < (unsigned)s)
      smem[REDOFF + threadIdx.x] += smem[REDOFF + threadIdx.x + s];
    __syncthreads();
  }
  const float out = smem[REDOFF];
  __syncthreads();
  return sqrtf(out);
}

// ---------------- matmul-only spectral primitives ----------------
// Coupled Newton-Schulz: outS = A^(1/2), outIS = A^(-1/2). outS may alias aO.
// Pass sO/isO = -1 to skip an output.
__device__ void ns_sqrt_invsqrt(int aO, int sO, int isO, int Y, int Z, int T, int W) {
  float c = frob(aO);
  c = fmaxf(c, 1e-20f);
  ew_scale_add_I(Y, aO, 1.f / c, 0.f);    // Y0 = A/c  (spectrum in (0,1])
  ew_setI(Z);                              // Z0 = I
  for (int it = 0; it < NS_ITERS; ++it) {
    mm64(T, Z, Y);
    ew_scale_add_I(T, T, -0.5f, 1.5f);     // T = (3I - ZY)/2
    mm64(W, Y, T); { int t = Y; Y = W; W = t; }   // Y = Y T
    mm64(W, T, Z); { int t = Z; Z = W; W = t; }   // Z = T Z
  }
  const float sc = sqrtf(c);
  if (sO  >= 0) ew_scale_add_I(sO,  Y, sc, 0.f);
  if (isO >= 0) ew_scale_add_I(isO, Z, 1.f / sc, 0.f);
}

// ACC += coeff * logm(X). X destroyed. Inverse scaling-and-squaring + Gregory series.
__device__ void logm_acc(int accO, float coeff, int X, int s1, int s2, int s3, int s4) {
  for (int i = 0; i < LOG_SQRTS; ++i)
    ns_sqrt_invsqrt(X, X, -1, s1, s2, s3, s4);   // X <- X^(1/16)
  ew_scale_add_I(X, X, 1.f, -1.f);               // M = X - I (small)
  ew_copy(s1, X);                                // P = M
  const float cc = coeff * (float)(1 << LOG_SQRTS);
  float sign = 1.f;
  for (int j = 1; j <= LOG_TERMS; ++j) {
    ew_axpy(accO, s1, sign * cc / (float)j);     // ACC += cc*(-1)^{j+1} M^j / j
    sign = -sign;
    if (j < LOG_TERMS) { mm64(s2, s1, X); int t = s1; s1 = s2; s2 = t; }
  }
}

// E = expm(X) via scaling-and-squaring (scale 1/64, 8 Taylor terms, 6 squarings).
__device__ void expm(int E, int X, int s1, int s2, int s3) {
  ew_scale_add_I(s1, X, 1.f / 64.f, 0.f);   // T = X/64
  ew_copy(s2, s1);                           // power = T
  ew_setI(E);
  float f = 1.f;
  for (int j = 1; j <= EXP_TERMS; ++j) {
    f /= (float)j;
    ew_axpy(E, s2, f);
    if (j < EXP_TERMS) { mm64(s3, s2, s1); int t = s2; s2 = s3; s3 = t; }
  }
  for (int i = 0; i < EXP_SQUARES; ++i) { mm64(s3, E, E); ew_copy(E, s3); }
}

// buffer offsets inside the LDS pool
#define B0 (0 * MSZ)
#define B1 (1 * MSZ)
#define B2 (2 * MSZ)
#define B3 (3 * MSZ)
#define B4 (4 * MSZ)
#define B5 (5 * MSZ)
#define B6 (6 * MSZ)
#define B7 (7 * MSZ)

// ---------------- kernels ----------------
// Weighted Riemannian barycenter pooling over channel pairs (one block per group).
__global__ void pool_kernel(const float* __restrict__ x,
                            const float* __restrict__ w1,
                            float* __restrict__ ws) {
  extern __shared__ float smem[];
  const int g  = blockIdx.x;          // 0..2047 = (batch, c_out)
  const int bi = g >> 3, co = g & 7;
  const float* X0 = x + (size_t)(bi * 16 + 2 * co)     * 4096;
  const float* X1 = x + (size_t)(bi * 16 + 2 * co + 1) * 4096;

  const float ew0 = expf(w1[0]), ew1 = expf(w1[1]);
  const float w0 = ew0 / (ew0 + ew1), w_1 = ew1 / (ew0 + ew1);

  // G = w0*X0 + w1*X1  (weighted arithmetic init)
  {
    const int e0 = threadIdx.x * 8;
    const int r = e0 >> 6, c0 = e0 & 63;
#pragma unroll
    for (int j = 0; j < 8; ++j)
      smem[B3 + r * TS + c0 + j] = w0 * X0[e0 + j] + w_1 * X1[e0 + j];
  }
  __syncthreads();
  ns_sqrt_invsqrt(B3, B0, B1, B4, B5, B6, B7);   // Gs=B0, Gis=B1
  ew_zero(B2);                                    // GT accumulator
  for (int f = 0; f < 2; ++f) {
    ew_load(B4, f ? X1 : X0);
    mm64(B5, B1, B4);        // Gis * Xf
    mm64(B3, B5, B1);        // (Gis Xf) Gis
    logm_acc(B2, f ? w_1 : w0, B3, B4, B5, B6, B7);
  }
  expm(B3, B2, B4, B5, B6);  // expm(GT)
  mm64(B4, B0, B3);
  mm64(B5, B4, B0);          // pooled = Gs expm(GT) Gs
  ew_store(ws + POOLED_OFF + (size_t)g * 4096, B5);
}

// dst[e] = mean_m src[m*4096 + e] (deterministic serial sum per element)
__global__ void reduce_mean_kernel(const float* __restrict__ src,
                                   float* __restrict__ dst, int count) {
  const int e = blockIdx.x * blockDim.x + threadIdx.x;
  if (e >= 4096) return;
  float s = 0.f;
  for (int m = 0; m < count; ++m) s += src[(size_t)m * 4096 + e];
  dst[e] = s / (float)count;
}

// Gs/Gis of the arithmetic batch mean (single block).
__global__ void mean_sqrt_kernel(float* __restrict__ ws) {
  ew_load(B0, ws + G_OFF);
  ns_sqrt_invsqrt(B0, B1, B2, B3, B4, B5, B6);
  ew_store(ws + GS_OFF,  B1);
  ew_store(ws + GIS_OFF, B2);
}

// L_i = logm(Gis X_i Gis) for the Karcher flow step (one block per matrix).
__global__ void logs_kernel(float* __restrict__ ws) {
  const int m = blockIdx.x;
  ew_load(B0, ws + GIS_OFF);
  ew_load(B1, ws + POOLED_OFF + (size_t)m * 4096);
  mm64(B2, B0, B1);          // Gis * X
  mm64(B1, B2, B0);          // (Gis X) Gis
  ew_zero(B3);
  logm_acc(B3, 1.f, B1, B4, B5, B6, B7);
  ew_store(ws + LOGS_OFF + (size_t)m * 4096, B3);
}

// mean = Gs expm(GT) Gs; Mis = mean^(-1/2); Wsq = bn_weight^(1/2) (single block).
__global__ void finalize_kernel(float* __restrict__ ws, const float* __restrict__ bnw) {
  ew_load(B0, ws + GT_OFF);
  ew_load(B1, ws + GS_OFF);
  expm(B2, B0, B3, B4, B5);
  mm64(B3, B1, B2);
  mm64(B4, B3, B1);                              // batch mean
  ns_sqrt_invsqrt(B4, B0, B2, B3, B5, B6, B7);   // Mis = B2
  ew_store(ws + MIS_OFF, B2);
  ew_load(B0, bnw);
  ns_sqrt_invsqrt(B0, B1, B3, B4, B5, B6, B7);   // Wsq = B1
  ew_store(ws + WSQ_OFF, B1);
}

// centered/normed congruences + ReEig: out = (N + eps I + |N - eps I|)/2.
__global__ void final_kernel(float* __restrict__ ws, float* __restrict__ out) {
  extern __shared__ float smem[];
  const int m = blockIdx.x;
  ew_load(B0, ws + MIS_OFF);
  ew_load(B1, ws + WSQ_OFF);
  ew_load(B2, ws + POOLED_OFF + (size_t)m * 4096);
  mm64(B4, B0, B2);
  mm64(B2, B4, B0);          // centered = Mis X Mis
  mm64(B4, B1, B2);
  mm64(B3, B4, B1);          // N = Ws centered Ws
  ew_scale_add_I(B2, B3, 1.f, -EPS_REEIG);       // A = N - eps I
  mm64(B4, B2, B2);                              // A^2 (PSD)
  ns_sqrt_invsqrt(B4, B4, -1, B5, B6, B7, B2);   // |A| in B4
  float* o = out + (size_t)m * 4096;
  {
    const int e0 = threadIdx.x * 8;
    const int r = e0 >> 6, c0 = e0 & 63;
#pragma unroll
    for (int j = 0; j < 8; ++j) {
      const int c = c0 + j;
      o[e0 + j] = 0.5f * (smem[B3 + r * TS + c] + smem[B4 + r * TS + c] +
                          ((r == c) ? EPS_REEIG : 0.f));
    }
  }
}

// ---------------- launch ----------------
extern "C" void kernel_launch(void* const* d_in, const int* in_sizes, int n_in,
                              void* d_out, int out_size, void* d_ws, size_t ws_size,
                              hipStream_t stream) {
  (void)in_sizes; (void)n_in; (void)out_size; (void)ws_size;
  const float* x   = (const float*)d_in[0];   // [256,16,64,64]
  const float* w1  = (const float*)d_in[1];   // [2]
  const float* bnw = (const float*)d_in[2];   // [64,64]
  float* ws  = (float*)d_ws;
  float* out = (float*)d_out;                 // [256,8,64,64]

  const size_t shbytes = (size_t)POOL_FLOATS * sizeof(float);  // ~152 KB (<320 KB WGP)
  dim3 blk(NTHREADS);

  pool_kernel<<<dim3(2048), blk, shbytes, stream>>>(x, w1, ws);
  reduce_mean_kernel<<<dim3(8), blk, 0, stream>>>(ws + POOLED_OFF, ws + G_OFF, 2048);
  mean_sqrt_kernel<<<dim3(1), blk, shbytes, stream>>>(ws);
  logs_kernel<<<dim3(2048), blk, shbytes, stream>>>(ws);
  reduce_mean_kernel<<<dim3(8), blk, 0, stream>>>(ws + LOGS_OFF, ws + GT_OFF, 2048);
  finalize_kernel<<<dim3(1), blk, shbytes, stream>>>(ws, bnw);
  final_kernel<<<dim3(2048), blk, shbytes, stream>>>(ws, out);
}
